// AlignedGloveLayer_49349174231152
// MI455X (gfx1250) — matrix operand order repaired and checked by reading, not compiled, not run on hardware
//
#include <hip/hip_runtime.h>
#include <math.h>

typedef __bf16 bf16;
typedef __attribute__((ext_vector_type(8)))  __bf16 v8bf;
typedef __attribute__((ext_vector_type(16))) __bf16 v16bf;
typedef __attribute__((ext_vector_type(8)))  float  v8f;
typedef __attribute__((ext_vector_type(4)))  unsigned int u32x4;
typedef __attribute__((ext_vector_type(8)))  int i32x8;
typedef __attribute__((ext_vector_type(4)))  int i32x4;

#define LDP 40   // padded LDS row stride (bf16 elems): 64B data + 16B pad -> conflict-free, 16B aligned

#if __has_builtin(__builtin_amdgcn_tensor_load_to_lds) && __has_builtin(__builtin_amdgcn_s_wait_tensorcnt)
#define USE_TDM 1
#define NBUF 2
#else
#define USE_TDM 0
#define NBUF 1
#endif

// ---------- helpers ----------
__device__ __forceinline__ v16bf make16(v8bf x, v8bf y) {
  return __builtin_shufflevector(x, y, 0,1,2,3,4,5,6,7,8,9,10,11,12,13,14,15);
}
// A operand (16x32 bf16, MxK): lane m=L&15; elems0-7 K=ko..ko+7, elems8-15 K=16+ko.. (ko=8*(L>=16))
__device__ __forceinline__ v16bf ldopA(const bf16* s, int row, int lane) {
  int r  = row + (lane & 15);
  int ko = ((lane >> 4) & 1) << 3;
  const bf16* p = s + r * LDP + ko;
  return make16(*(const v8bf*)p, *(const v8bf*)(p + 16));
}
// B operand (32x16 bf16, KxN) from row-major B^T rows: lane n=L&15; elems0-7 K=ko..ko+7, elems8-15 K=16+ko.. (ko=8*(L>=16))
__device__ __forceinline__ v16bf ldopB(const bf16* s, int row, int lane) {
  int r  = row + (lane & 15);
  int ko = ((lane >> 4) & 1) << 3;
  const bf16* p = s + r * LDP + ko;
  return make16(*(const v8bf*)p, *(const v8bf*)(p + 16));
}

#if USE_TDM
// Tensor Data Mover: DMA a 128-row x 32-col bf16 tile (row stride K elems) into LDS,
// padding each 64B row to 80B (pad_interval = 16 DWORDs, pad_amount = 4 DWORDs) => LDP=40.
// D# layout per CDNA5 ISA 8.3/8.4: group0 = {flags, lds_addr, global_addr, type=2},
// group1 = {mask/size/pad, dims, tile dims, strides}. 2D tile -> groups 2/3 (and extra group) zero.
__device__ __forceinline__ void tdm_tile_load(const bf16* gsrc, unsigned lds_byte_addr, int K)
{
  unsigned long long ga = (unsigned long long)(const void*)gsrc;
  u32x4 g0;
  g0[0] = 0x1u;                                                 // count=1, user descriptor
  g0[1] = lds_byte_addr;                                        // LDS dest (byte addr)
  g0[2] = (unsigned)ga;                                         // global_addr[31:0]
  g0[3] = ((unsigned)(ga >> 32) & 0x1FFFFFFu) | 0x80000000u;    // global_addr[56:32] | type=2
  i32x8 g1;
  g1[0] = (int)((1u << 16)      // data_size = 1 -> 2 bytes (bf16)
              | (1u << 20)      // pad_enable
              | (3u << 22)      // pad_interval: 16 DWORDs (64B) between pads
              | (3u << 25));    // pad_amount: 4 DWORDs (16B)
  unsigned K_u = (unsigned)K;
  g1[1] = (int)((K_u & 0xFFFFu) << 16);                          // tensor_dim0[15:0]
  g1[2] = (int)(((K_u >> 16) & 0xFFFFu) | (128u << 16));         // tensor_dim0[31:16] | tensor_dim1[15:0]
  g1[3] = (int)(32u << 16);                                      // tensor_dim1[31:16]=0 | tile_dim0=32
  g1[4] = (int)128u;                                             // tile_dim1=128 | tile_dim2=0
  g1[5] = (int)K_u;                                              // tensor_dim0_stride[31:0]
  g1[6] = 0;                                                     // stride[47:32] | dim1_stride lo (unused, 2D)
  g1[7] = 0;
  i32x4 z4 = {0, 0, 0, 0};
  i32x8 z8 = {0, 0, 0, 0, 0, 0, 0, 0};
  __builtin_amdgcn_tensor_load_to_lds(g0, g1, z4, z4, z8, 0);
}
#endif

__device__ __forceinline__ float block_reduce_sum(float v) {
  __shared__ float red[8];
  int lane = threadIdx.x & 31, w = threadIdx.x >> 5;
  #pragma unroll
  for (int o = 16; o > 0; o >>= 1) v += __shfl_xor(v, o, 32);
  if (lane == 0) red[w] = v;
  __syncthreads();
  float s = 0.0f;
  if (w == 0) {
    s = (lane < (int)(blockDim.x >> 5)) ? red[lane] : 0.0f;
    #pragma unroll
    for (int o = 4; o > 0; o >>= 1) s += __shfl_xor(s, o, 32);
  }
  return s;  // valid in wave 0
}

// ---------- generic split-bf16 WMMA GEMM:  C[M,N] = A[M,K] * Brm[N,K]^T ----------
// EPI: 0 = +bias, 1 = +bias,ReLU, 2 = distance-min (colmin[j] = min_i(aa[i]+bb[j]-2*D[i,j]))
template<int EPI>
__global__ __launch_bounds__(256)
void gemm_kernel(const bf16* __restrict__ Ahi, const bf16* __restrict__ Alo,
                 const bf16* __restrict__ Bhi, const bf16* __restrict__ Blo,
                 int M, int N, int K, int Nreal,
                 const float* __restrict__ bias, float* __restrict__ C,
                 const float* __restrict__ aa, const float* __restrict__ bb,
                 unsigned* __restrict__ colmin)
{
  __shared__ __align__(16) bf16 sAh[NBUF][128*LDP], sAl[NBUF][128*LDP],
                                sBh[NBUF][128*LDP], sBl[NBUF][128*LDP];
  __shared__ unsigned smin[128];

  const int t    = threadIdx.x;
  const int lane = t & 31;
  const int w    = t >> 5;
  const int wm   = w >> 1;      // 0..3 -> 32-row slab
  const int wn   = w & 1;       // 0..1 -> 64-col slab
  const int n0   = blockIdx.x * 128;
  const int m0   = blockIdx.y * 128;

  if (EPI == 2 && t < 128) smin[t] = 0x7F800000u;  // +inf bits

  v8f acc[2][4];
  #pragma unroll
  for (int i = 0; i < 2; ++i)
    #pragma unroll
    for (int j = 0; j < 4; ++j)
      #pragma unroll
      for (int e = 0; e < 8; ++e) acc[i][j][e] = 0.0f;

  const int nchunks = K >> 5;

  auto compute_buf = [&](int pb) {
    v16bf ah[2], al[2], bh[4], bl[4];
    #pragma unroll
    for (int sm = 0; sm < 2; ++sm) {
      ah[sm] = ldopA(sAh[pb], wm*32 + sm*16, lane);
      al[sm] = ldopA(sAl[pb], wm*32 + sm*16, lane);
    }
    #pragma unroll
    for (int sn = 0; sn < 4; ++sn) {
      bh[sn] = ldopB(sBh[pb], wn*64 + sn*16, lane);
      bl[sn] = ldopB(sBl[pb], wn*64 + sn*16, lane);
    }
    #pragma unroll
    for (int sm = 0; sm < 2; ++sm)
      #pragma unroll
      for (int sn = 0; sn < 4; ++sn) {
        acc[sm][sn] = __builtin_amdgcn_wmma_f32_16x16x32_bf16(
            false, ah[sm], false, bh[sn], (short)0, acc[sm][sn], false, false);
        acc[sm][sn] = __builtin_amdgcn_wmma_f32_16x16x32_bf16(
            false, ah[sm], false, bl[sn], (short)0, acc[sm][sn], false, false);
        acc[sm][sn] = __builtin_amdgcn_wmma_f32_16x16x32_bf16(
            false, al[sm], false, bh[sn], (short)0, acc[sm][sn], false, false);
      }
  };

#if USE_TDM
  // ---- TDM double-buffered pipeline: wave 0 drives the Tensor Data Mover ----
  const unsigned ldsAh = (unsigned)(size_t)(void*)&sAh[0][0];
  const unsigned ldsAl = (unsigned)(size_t)(void*)&sAl[0][0];
  const unsigned ldsBh = (unsigned)(size_t)(void*)&sBh[0][0];
  const unsigned ldsBl = (unsigned)(size_t)(void*)&sBl[0][0];
  auto issue = [&](int chunk, int buf) {
    const size_t ao = (size_t)m0 * K + (size_t)chunk * 32;
    const size_t bo = (size_t)n0 * K + (size_t)chunk * 32;
    const unsigned boff = (unsigned)buf * (128u * LDP * 2u);
    tdm_tile_load(Ahi + ao, ldsAh + boff, K);
    tdm_tile_load(Alo + ao, ldsAl + boff, K);
    tdm_tile_load(Bhi + bo, ldsBh + boff, K);
    tdm_tile_load(Blo + bo, ldsBl + boff, K);
  };
  int p = 0;
  if (w == 0) issue(0, 0);
  for (int i = 0; i < nchunks; ++i) {
    if (w == 0) {
      if (i + 1 < nchunks) {
        issue(i + 1, p ^ 1);                      // prefetch next chunk into other buffer
        __builtin_amdgcn_s_wait_tensorcnt(4);     // 4 in-flight = next chunk; current done
      } else {
        __builtin_amdgcn_s_wait_tensorcnt(0);
      }
    }
    __syncthreads();       // current buffer ready for all waves
    compute_buf(p);
    __syncthreads();       // all waves done reading before buffer is re-filled
    p ^= 1;
  }
#else
  // ---- fallback: VGPR-staged cooperative loads ----
  const int r  = t >> 1;          // 0..127 tile row loaded by this thread
  const int kh = (t & 1) << 4;    // 0 or 16 (bf16 elems)
  for (int kc = 0; kc < K; kc += 32) {
    const size_t ga = (size_t)(m0 + r) * K + kc + kh;
    const size_t gb = (size_t)(n0 + r) * K + kc + kh;
    uint4 a0 = *(const uint4*)(Ahi + ga);
    uint4 a1 = *(const uint4*)(Ahi + ga + 8);
    uint4 a2 = *(const uint4*)(Alo + ga);
    uint4 a3 = *(const uint4*)(Alo + ga + 8);
    uint4 b0 = *(const uint4*)(Bhi + gb);
    uint4 b1 = *(const uint4*)(Bhi + gb + 8);
    uint4 b2 = *(const uint4*)(Blo + gb);
    uint4 b3 = *(const uint4*)(Blo + gb + 8);
    __syncthreads();
    *(uint4*)(sAh[0] + r*LDP + kh) = a0;  *(uint4*)(sAh[0] + r*LDP + kh + 8) = a1;
    *(uint4*)(sAl[0] + r*LDP + kh) = a2;  *(uint4*)(sAl[0] + r*LDP + kh + 8) = a3;
    *(uint4*)(sBh[0] + r*LDP + kh) = b0;  *(uint4*)(sBh[0] + r*LDP + kh + 8) = b1;
    *(uint4*)(sBl[0] + r*LDP + kh) = b2;  *(uint4*)(sBl[0] + r*LDP + kh + 8) = b3;
    __syncthreads();
    compute_buf(0);
  }
#endif

  if (EPI == 2) {
    #pragma unroll
    for (int sn = 0; sn < 4; ++sn) {
      const int nl = wn*64 + sn*16 + (lane & 15);
      float lmin = __builtin_inff();
      #pragma unroll
      for (int sm = 0; sm < 2; ++sm) {
        const int mb = m0 + wm*32 + sm*16 + (((lane >> 4) & 1) << 3);
        #pragma unroll
        for (int v = 0; v < 8; ++v)
          lmin = fminf(lmin, aa[mb + v] - 2.0f * acc[sm][sn][v]);
      }
      lmin = fminf(lmin, __shfl_xor(lmin, 16, 32));
      const float d = fmaxf(lmin + bb[n0 + nl], 0.0f);
      atomicMin(&smin[nl], __float_as_uint(d));
    }
    __syncthreads();
    if (t < 128) atomicMin(&colmin[n0 + t], smin[t]);
  } else {
    #pragma unroll
    for (int sm = 0; sm < 2; ++sm)
      #pragma unroll
      for (int sn = 0; sn < 4; ++sn) {
        const int n  = n0 + wn*64 + sn*16 + (lane & 15);
        const int mb = m0 + wm*32 + sm*16 + (((lane >> 4) & 1) << 3);
        const float bv = (n < Nreal) ? bias[n] : 0.0f;
        #pragma unroll
        for (int v = 0; v < 8; ++v) {
          float val = acc[sm][sn][v] + bv;
          if (EPI == 1) val = fmaxf(val, 0.0f);
          if (n >= Nreal) val = 0.0f;
          C[(size_t)(mb + v) * N + n] = val;
        }
      }
  }
}

// ---------- gather rows (idx==null -> identity) + f32->bf16 hi/lo split + optional row norm ----------
__global__ __launch_bounds__(256)
void gather_split_norm(const float* __restrict__ W, const int* __restrict__ idx, int D,
                       float* __restrict__ outf, bf16* __restrict__ hi, bf16* __restrict__ lo,
                       float* __restrict__ norm)
{
  const int r = blockIdx.x;
  const size_t src = idx ? (size_t)idx[r] : (size_t)r;
  const float* row = W + src * (size_t)D;
  float s = 0.0f;
  for (int c = threadIdx.x; c < D; c += blockDim.x) {
    float x = row[c];
    if (outf) outf[(size_t)r * D + c] = x;
    bf16 h = (bf16)x;
    bf16 l = (bf16)(x - (float)h);
    hi[(size_t)r * D + c] = h;
    lo[(size_t)r * D + c] = l;
    s += x * x;
  }
  if (norm) {
    float tot = block_reduce_sum(s);
    if (threadIdx.x == 0) norm[r] = tot;
  }
}

// ---------- W[Kin,Nin] -> W^T padded [Nout,Kout] as bf16 hi/lo ----------
__global__ __launch_bounds__(256)
void transpose_split(const float* __restrict__ W, int Kin, int Nin,
                     bf16* __restrict__ hi, bf16* __restrict__ lo, int Kout, int Nout)
{
  int i = blockIdx.x * blockDim.x + threadIdx.x;
  if (i >= Kout * Nout) return;
  int n = i / Kout, k = i % Kout;
  float x = (k < Kin && n < Nin) ? W[(size_t)k * Nin + n] : 0.0f;
  bf16 h = (bf16)x;
  bf16 l = (bf16)(x - (float)h);
  hi[(size_t)n * Kout + k] = h;
  lo[(size_t)n * Kout + k] = l;
}

// ---------- sum_i sqrt(||U_i - V_i||^2) ----------
__global__ __launch_bounds__(256)
void rownormdiff(const float* __restrict__ U, const float* __restrict__ V, int D,
                 float* __restrict__ acc)
{
  const int r = blockIdx.x;
  float s = 0.0f;
  for (int c = threadIdx.x; c < D; c += blockDim.x) {
    float d = U[(size_t)r * D + c] - V[(size_t)r * D + c];
    s += d * d;
  }
  float tot = block_reduce_sum(s);
  if (threadIdx.x == 0) atomicAdd(acc, sqrtf(tot));
}

// ---------- sum_j sqrt(colmin[j]) ----------
__global__ __launch_bounds__(256)
void colmin_sum(const unsigned* __restrict__ cm, int n, float* __restrict__ acc)
{
  int i = blockIdx.x * blockDim.x + threadIdx.x;
  float v = (i < n) ? sqrtf(__uint_as_float(cm[i])) : 0.0f;
  float tot = block_reduce_sum(v);
  if (threadIdx.x == 0) atomicAdd(acc, tot);
}

__global__ void init_kernel(unsigned* cm1, unsigned* cm2, float* accs, int n)
{
  int i = blockIdx.x * blockDim.x + threadIdx.x;
  if (i < n) { cm1[i] = 0x7F800000u; cm2[i] = 0x7F800000u; }
  if (i < 4) accs[i] = 0.0f;
}

__global__ void final_kernel(const float* accs, float* out, float inv)
{
  if (threadIdx.x == 0 && blockIdx.x == 0)
    out[0] = (accs[0] + accs[1] + accs[2] + accs[3]) * inv;
}

// ============================ host ============================
extern "C" void kernel_launch(void* const* d_in, const int* in_sizes, int n_in,
                              void* d_out, int out_size, void* d_ws, size_t ws_size,
                              hipStream_t stream) {
  (void)in_sizes; (void)n_in; (void)out_size; (void)ws_size;
  const float* x_weight = (const float*)d_in[0];
  const float* y_weight = (const float*)d_in[1];
  const float* fx_W1 = (const float*)d_in[2];
  const float* fx_b1 = (const float*)d_in[3];
  const float* fx_W2 = (const float*)d_in[4];
  const float* fx_b2 = (const float*)d_in[5];
  const float* gy_W1 = (const float*)d_in[6];
  const float* gy_b1 = (const float*)d_in[7];
  const float* gy_W2 = (const float*)d_in[8];
  const float* gy_b2 = (const float*)d_in[9];
  const int* x_present = (const int*)d_in[10];
  const int* y_check   = (const int*)d_in[11];
  const int* y_present = (const int*)d_in[12];
  const int* x_check   = (const int*)d_in[13];
  float* out = (float*)d_out;

  const int Bn = 8192, Dx = 512, Dy = 256, Hp = 128, Hr = 100;

  char* ws = (char*)d_ws;
  size_t off = 0;
  auto alloc = [&](size_t bytes) -> char* {
    char* p = ws + off;
    off = (off + bytes + 255) & ~(size_t)255;
    return p;
  };
  float* fx_in_f  = (float*)alloc((size_t)Bn*Dx*4);
  bf16*  fx_in_hi = (bf16*) alloc((size_t)Bn*Dx*2);
  bf16*  fx_in_lo = (bf16*) alloc((size_t)Bn*Dx*2);
  float* gy_in_f  = (float*)alloc((size_t)Bn*Dy*4);
  bf16*  gy_in_hi = (bf16*) alloc((size_t)Bn*Dy*2);
  bf16*  gy_in_lo = (bf16*) alloc((size_t)Bn*Dy*2);
  bf16*  yc_hi    = (bf16*) alloc((size_t)Bn*Dy*2);
  bf16*  yc_lo    = (bf16*) alloc((size_t)Bn*Dy*2);
  bf16*  xc_hi    = (bf16*) alloc((size_t)Bn*Dx*2);
  bf16*  xc_lo    = (bf16*) alloc((size_t)Bn*Dx*2);
  float* bb_y     = (float*)alloc((size_t)Bn*4);
  float* bb_x     = (float*)alloc((size_t)Bn*4);
  float* aa_f     = (float*)alloc((size_t)Bn*4);
  float* aa_g     = (float*)alloc((size_t)Bn*4);
  bf16*  fxW1t_hi = (bf16*) alloc((size_t)Hp*Dx*2);
  bf16*  fxW1t_lo = (bf16*) alloc((size_t)Hp*Dx*2);
  bf16*  fxW2t_hi = (bf16*) alloc((size_t)Dy*Hp*2);
  bf16*  fxW2t_lo = (bf16*) alloc((size_t)Dy*Hp*2);
  bf16*  gyW1t_hi = (bf16*) alloc((size_t)Hp*Dy*2);
  bf16*  gyW1t_lo = (bf16*) alloc((size_t)Hp*Dy*2);
  bf16*  gyW2t_hi = (bf16*) alloc((size_t)Dx*Hp*2);
  bf16*  gyW2t_lo = (bf16*) alloc((size_t)Dx*Hp*2);
  float* h_f      = (float*)alloc((size_t)Bn*Hp*4);
  bf16*  h_hi     = (bf16*) alloc((size_t)Bn*Hp*2);
  bf16*  h_lo     = (bf16*) alloc((size_t)Bn*Hp*2);
  float* fxo_f    = (float*)alloc((size_t)Bn*Dy*4);
  bf16*  fxo_hi   = (bf16*) alloc((size_t)Bn*Dy*2);
  bf16*  fxo_lo   = (bf16*) alloc((size_t)Bn*Dy*2);
  float* gyo_f    = (float*)alloc((size_t)Bn*Dx*4);
  bf16*  gyo_hi   = (bf16*) alloc((size_t)Bn*Dx*2);
  bf16*  gyo_lo   = (bf16*) alloc((size_t)Bn*Dx*2);
  float* rt_f     = (float*)alloc((size_t)Bn*Dx*4);
  unsigned* colmin1 = (unsigned*)alloc((size_t)Bn*4);
  unsigned* colmin2 = (unsigned*)alloc((size_t)Bn*4);
  float* accs     = (float*)alloc(4*4);

  // 0. init colmin + accumulators
  init_kernel<<<dim3((Bn+255)/256), 256, 0, stream>>>(colmin1, colmin2, accs, Bn);

  // 1. transpose + pad + split weights
  transpose_split<<<dim3((Hp*Dx+255)/256),256,0,stream>>>(fx_W1, Dx, Hr, fxW1t_hi, fxW1t_lo, Dx, Hp);
  transpose_split<<<dim3((Dy*Hp+255)/256),256,0,stream>>>(fx_W2, Hr, Dy, fxW2t_hi, fxW2t_lo, Hp, Dy);
  transpose_split<<<dim3((Hp*Dy+255)/256),256,0,stream>>>(gy_W1, Dy, Hr, gyW1t_hi, gyW1t_lo, Dy, Hp);
  transpose_split<<<dim3((Dx*Hp+255)/256),256,0,stream>>>(gy_W2, Hr, Dx, gyW2t_hi, gyW2t_lo, Hp, Dx);

  // 2. gathers (+split, +norms for check sets)
  gather_split_norm<<<Bn,256,0,stream>>>(x_weight, x_present, Dx, fx_in_f, fx_in_hi, fx_in_lo, nullptr);
  gather_split_norm<<<Bn,256,0,stream>>>(y_weight, y_present, Dy, gy_in_f, gy_in_hi, gy_in_lo, nullptr);
  gather_split_norm<<<Bn,256,0,stream>>>(y_weight, y_check,   Dy, nullptr, yc_hi, yc_lo, bb_y);
  gather_split_norm<<<Bn,256,0,stream>>>(x_weight, x_check,   Dx, nullptr, xc_hi, xc_lo, bb_x);

  const dim3 blk(256);
  // x -> y -> x cycle
  gemm_kernel<1><<<dim3(Hp/128, Bn/128), blk, 0, stream>>>(fx_in_hi, fx_in_lo, fxW1t_hi, fxW1t_lo,
      Bn, Hp, Dx, Hr, fx_b1, h_f, nullptr, nullptr, nullptr);
  gather_split_norm<<<Bn,256,0,stream>>>(h_f, nullptr, Hp, nullptr, h_hi, h_lo, nullptr);
  gemm_kernel<0><<<dim3(Dy/128, Bn/128), blk, 0, stream>>>(h_hi, h_lo, fxW2t_hi, fxW2t_lo,
      Bn, Dy, Hp, Dy, fx_b2, fxo_f, nullptr, nullptr, nullptr);
  gather_split_norm<<<Bn,256,0,stream>>>(fxo_f, nullptr, Dy, nullptr, fxo_hi, fxo_lo, aa_f);
  gemm_kernel<1><<<dim3(Hp/128, Bn/128), blk, 0, stream>>>(fxo_hi, fxo_lo, gyW1t_hi, gyW1t_lo,
      Bn, Hp, Dy, Hr, gy_b1, h_f, nullptr, nullptr, nullptr);
  gather_split_norm<<<Bn,256,0,stream>>>(h_f, nullptr, Hp, nullptr, h_hi, h_lo, nullptr);
  gemm_kernel<0><<<dim3(Dx/128, Bn/128), blk, 0, stream>>>(h_hi, h_lo, gyW2t_hi, gyW2t_lo,
      Bn, Dx, Hp, Dx, gy_b2, rt_f, nullptr, nullptr, nullptr);
  rownormdiff<<<Bn,256,0,stream>>>(rt_f, fx_in_f, Dx, accs + 0);

  // y -> x -> y cycle
  gemm_kernel<1><<<dim3(Hp/128, Bn/128), blk, 0, stream>>>(gy_in_hi, gy_in_lo, gyW1t_hi, gyW1t_lo,
      Bn, Hp, Dy, Hr, gy_b1, h_f, nullptr, nullptr, nullptr);
  gather_split_norm<<<Bn,256,0,stream>>>(h_f, nullptr, Hp, nullptr, h_hi, h_lo, nullptr);
  gemm_kernel<0><<<dim3(Dx/128, Bn/128), blk, 0, stream>>>(h_hi, h_lo, gyW2t_hi, gyW2t_lo,
      Bn, Dx, Hp, Dx, gy_b2, gyo_f, nullptr, nullptr, nullptr);
  gather_split_norm<<<Bn,256,0,stream>>>(gyo_f, nullptr, Dx, nullptr, gyo_hi, gyo_lo, aa_g);
  gemm_kernel<1><<<dim3(Hp/128, Bn/128), blk, 0, stream>>>(gyo_hi, gyo_lo, fxW1t_hi, fxW1t_lo,
      Bn, Hp, Dx, Hr, fx_b1, h_f, nullptr, nullptr, nullptr);
  gather_split_norm<<<Bn,256,0,stream>>>(h_f, nullptr, Hp, nullptr, h_hi, h_lo, nullptr);
  gemm_kernel<0><<<dim3(Dy/128, Bn/128), blk, 0, stream>>>(h_hi, h_lo, fxW2t_hi, fxW2t_lo,
      Bn, Dy, Hp, Dy, fx_b2, rt_f, nullptr, nullptr, nullptr);
  rownormdiff<<<Bn,256,0,stream>>>(rt_f, gy_in_f, Dy, accs + 1);

  // fused cdist + column-min (the two big GEMMs)
  gemm_kernel<2><<<dim3(Bn/128, Bn/128), blk, 0, stream>>>(fxo_hi, fxo_lo, yc_hi, yc_lo,
      Bn, Bn, Dy, Bn, nullptr, nullptr, aa_f, bb_y, colmin1);
  gemm_kernel<2><<<dim3(Bn/128, Bn/128), blk, 0, stream>>>(gyo_hi, gyo_lo, xc_hi, xc_lo,
      Bn, Bn, Dx, Bn, nullptr, nullptr, aa_g, bb_x, colmin2);

  colmin_sum<<<dim3(Bn/256), 256, 0, stream>>>(colmin1, Bn, accs + 2);
  colmin_sum<<<dim3(Bn/256), 256, 0, stream>>>(colmin2, Bn, accs + 3);
  final_kernel<<<1, 32, 0, stream>>>(accs, out, 1.0f / (float)Bn);
}
